// RelativeGlobalAttention_56367150792993
// MI455X (gfx1250) — matrix operand order repaired and checked by reading, not compiled, not run on hardware
//
#include <hip/hip_runtime.h>
#include <hip/hip_bf16.h>

typedef _Float16 f16;
typedef __attribute__((ext_vector_type(16))) _Float16 v16h;
typedef __attribute__((ext_vector_type(8)))  _Float16 v8h;
typedef __attribute__((ext_vector_type(8)))  float    v8f;

#define B_   4
#define S_   4096
#define HID_ 128
#define H_   4
#define D_   32

// ---------------------------------------------------------------------------
// Kernel 0: convert Er [S,D] fp32 -> f16 (scale + log2e are folded into q)
// ---------------------------------------------------------------------------
__global__ void rga_er_cvt(const float* __restrict__ Er, f16* __restrict__ er16, int n) {
    int i = blockIdx.x * blockDim.x + threadIdx.x;
    if (i < n) er16[i] = (f16)Er[i];
}

// ---------------------------------------------------------------------------
// Kernel 1: QKV projection (fp32 accumulate), emit:
//   q16  [B,H,S,D] f16, pre-scaled by log2(e)/sqrt(D)  (softmax in exp2 domain)
//   k16  [B,H,S,D] f16
//   vt16 [B,H,D,S] f16  (V transposed so it loads as a contiguous WMMA B operand)
// ---------------------------------------------------------------------------
__global__ __launch_bounds__(128)
void rga_qkv_proj(const float* __restrict__ x,
                  const float* __restrict__ Wq, const float* __restrict__ bq,
                  const float* __restrict__ Wk, const float* __restrict__ bk,
                  const float* __restrict__ Wv, const float* __restrict__ bv,
                  f16* __restrict__ q16, f16* __restrict__ k16, f16* __restrict__ vt16)
{
    __shared__ float xs[8][HID_];
    const int c  = threadIdx.x;           // output column 0..127
    const int g0 = blockIdx.x * 8;        // first global row (b*S + s)
    for (int r = 0; r < 8; ++r) xs[r][c] = x[(size_t)(g0 + r) * HID_ + c];
    __syncthreads();

    float aq[8], ak[8], av[8];
    #pragma unroll
    for (int r = 0; r < 8; ++r) { aq[r] = 0.f; ak[r] = 0.f; av[r] = 0.f; }

    #pragma unroll 4
    for (int k = 0; k < HID_; ++k) {
        float wq = Wq[(size_t)k * HID_ + c];
        float wk = Wk[(size_t)k * HID_ + c];
        float wv = Wv[(size_t)k * HID_ + c];
        #pragma unroll
        for (int r = 0; r < 8; ++r) {
            float xv = xs[r][k];
            aq[r] = fmaf(xv, wq, aq[r]);
            ak[r] = fmaf(xv, wk, ak[r]);
            av[r] = fmaf(xv, wv, av[r]);
        }
    }

    const int   h = c >> 5, d = c & 31;
    const float bqc = bq[c], bkc = bk[c], bvc = bv[c];
    // log2(e) / sqrt(32): softmax computed with raw v_exp_f32 (2^x)
    const float qscale = 1.4426950408889634f * 0.17677669529663687f;
    #pragma unroll
    for (int r = 0; r < 8; ++r) {
        int g = g0 + r;
        int b = g / S_, s = g % S_;
        size_t bh = (size_t)(b * H_ + h);
        q16 [(bh * S_ + s) * D_ + d] = (f16)((aq[r] + bqc) * qscale);
        k16 [(bh * S_ + s) * D_ + d] = (f16)(ak[r] + bkc);
        vt16[(bh * D_ + d) * S_ + s] = (f16)(av[r] + bvc);
    }
}

// ---------------------------------------------------------------------------
// Kernel 2: fused flash attention with relative (skewed) logits.
// One wave per 16-query tile; 64-key chunks: 14 v_wmma_f32_16x16x32_f16 each
// (4x QK^T, 4x G=Q*Er^T for Srel, 4x P*V, 2x rowsum(P) via all-ones B).
// ---------------------------------------------------------------------------
__global__ __launch_bounds__(256)
void rga_flash(const f16* __restrict__ q16, const f16* __restrict__ k16,
               const f16* __restrict__ vt16, const f16* __restrict__ er16,
               float* __restrict__ out)
{
    __shared__ f16 lds_p[8][16 * 64];     // per-wave P buffer [query][key], 2KB/wave

    const int lane = threadIdx.x & 31;
    const int wid  = threadIdx.x >> 5;
    const int wg   = blockIdx.x * 8 + wid;     // global wave id == query tile id
    const int bh   = wg >> 8;                  // / (S/16)
    const int i0   = (wg & 255) << 4;          // query tile start
    const int n    = lane & 15;
    const int half = lane >> 4;

    const f16* qb = q16  + (size_t)bh * S_ * D_;
    const f16* kb = k16  + (size_t)bh * S_ * D_;
    const f16* vb = vt16 + (size_t)bh * D_ * S_;
    f16* pbuf = lds_p[wid];

    // --- A operand: Q tile (16x32), lane holds row m=n, two 16B chunks of d
    v16h a_q;
    {
        const f16* qrow = qb + (size_t)(i0 + n) * D_;
        v8h lo = *(const v8h*)(qrow + 16 * half);
        v8h hi = *(const v8h*)(qrow + 16 * half + 8);
        #pragma unroll
        for (int i = 0; i < 8; ++i) { a_q[i] = lo[i]; a_q[8 + i] = hi[i]; }
    }

    // --- all-ones B operand (row-sum trick: every output column = row sum)
    v16h b_ones;
    #pragma unroll
    for (int i = 0; i < 16; ++i) b_ones[i] = (f16)1.0f;

    // --- initial G_prev: Er rows rbase = (S-1-i0) - 16  (clamped; OOB cols unused)
    v8f g_prev;
    {
        int row = (S_ - 1 - i0 - 16) + n;
        row = row < 0 ? 0 : (row >= S_ ? S_ - 1 : row);
        v16h b_er = *(const v16h*)(er16 + (size_t)row * D_ + 16 * half);
        v8f z = {};
        g_prev = __builtin_amdgcn_wmma_f32_16x16x32_f16(false, a_q, false, b_er,
                                                        (short)0, z, false, false);
    }

    float mstat[8], lstat[8];
    #pragma unroll
    for (int r = 0; r < 8; ++r) { mstat[r] = -1e30f; lstat[r] = 0.f; }
    v8f o0 = {}, o1 = {};

    for (int j0 = 0; j0 <= i0; j0 += 64) {
        v8f s[4];
        #pragma unroll
        for (int t = 0; t < 4; ++t) {
            int j0h = j0 + 16 * t;
            if (j0h <= i0) {                       // wave-uniform branch
                // B operand: K^T tile (lane = key col, 32B contiguous d-range)
                v16h b_k = *(const v16h*)(kb + (size_t)(j0h + n) * D_ + 16 * half);
                v8f z = {};
                v8f sc = __builtin_amdgcn_wmma_f32_16x16x32_f16(false, a_q, false, b_k,
                                                                (short)0, z, false, false);
                // B operand: Er^T tile, rbase = S-1 + j0h - i0
                int row = (S_ - 1 + j0h - i0) + n;
                row = row < 0 ? 0 : (row >= S_ ? S_ - 1 : row);
                v16h b_er = *(const v16h*)(er16 + (size_t)row * D_ + 16 * half);
                v8f g_cur = __builtin_amdgcn_wmma_f32_16x16x32_f16(false, a_q, false, b_er,
                                                                   (short)0, z, false, false);
                // Srel diagonal gather: source-side cur/prev merge (u>=0 <=> c+m<=15),
                // then a single ds_bpermute per row.
                #pragma unroll
                for (int r = 0; r < 8; ++r) {
                    int   m      = 8 * half + r;
                    float merged = (n + m <= 15) ? g_cur[r] : g_prev[r];
                    int   src    = ((n - m + 16) & 15) + 16 * half;
                    sc[r] += __shfl(merged, src);
                }
                if (j0h == i0) {                   // causal mask: diagonal tile only
                    #pragma unroll
                    for (int r = 0; r < 8; ++r)
                        if (n > 8 * half + r) sc[r] = -1e30f;
                }
                g_prev = g_cur;
                s[t] = sc;
            } else {
                #pragma unroll
                for (int r = 0; r < 8; ++r) s[t][r] = -1e30f;
            }
        }

        // --- online softmax over this 64-key chunk (exp2 domain)
        float alpha[8];
        #pragma unroll
        for (int r = 0; r < 8; ++r) {
            float v = fmaxf(fmaxf(s[0][r], s[1][r]), fmaxf(s[2][r], s[3][r]));
            v = fmaxf(v, __shfl_xor(v, 1));
            v = fmaxf(v, __shfl_xor(v, 2));
            v = fmaxf(v, __shfl_xor(v, 4));
            v = fmaxf(v, __shfl_xor(v, 8));
            float mn = fmaxf(mstat[r], v);
            alpha[r] = __builtin_amdgcn_exp2f(mstat[r] - mn);
            mstat[r] = mn;
            #pragma unroll
            for (int t = 0; t < 4; ++t)
                s[t][r] = __builtin_amdgcn_exp2f(s[t][r] - mn);
        }

        // --- stage P as f16 in LDS [query][key0..63] (wave-private, DS in-order)
        #pragma unroll
        for (int t = 0; t < 4; ++t)
            #pragma unroll
            for (int r = 0; r < 8; ++r)
                pbuf[(8 * half + r) * 64 + 16 * t + n] = (f16)s[t][r];

        // --- A operands: P groups (16 queries x 32 keys), ds_load_b128 pairs
        v16h a_p0, a_p1;
        {
            v8h lo0 = *(const v8h*)&pbuf[n * 64 + 16 * half];
            v8h hi0 = *(const v8h*)&pbuf[n * 64 + 16 * half + 8];
            v8h lo1 = *(const v8h*)&pbuf[n * 64 + 32 + 16 * half];
            v8h hi1 = *(const v8h*)&pbuf[n * 64 + 32 + 16 * half + 8];
            #pragma unroll
            for (int i = 0; i < 8; ++i) {
                a_p0[i] = lo0[i]; a_p0[8 + i] = hi0[i];
                a_p1[i] = lo1[i]; a_p1[8 + i] = hi1[i];
            }
        }

        // --- row sums of P via WMMA against all-ones B (lands per-lane, no shuffles)
        {
            v8f zz = {};
            v8f rs = __builtin_amdgcn_wmma_f32_16x16x32_f16(false, a_p0, false, b_ones,
                                                            (short)0, zz, false, false);
            rs = __builtin_amdgcn_wmma_f32_16x16x32_f16(false, a_p1, false, b_ones,
                                                        (short)0, rs, false, false);
            #pragma unroll
            for (int r = 0; r < 8; ++r) lstat[r] = lstat[r] * alpha[r] + rs[r];
        }

        // --- B operands: V^T rows, contiguous 32B per lane (clamped vs padding)
        int jv0 = j0 + 16 * half;        if (jv0 > S_ - 16) jv0 = S_ - 16;
        int jv1 = j0 + 32 + 16 * half;   if (jv1 > S_ - 16) jv1 = S_ - 16;
        v16h b_v00 = *(const v16h*)(vb + (size_t)n        * S_ + jv0);
        v16h b_v01 = *(const v16h*)(vb + (size_t)n        * S_ + jv1);
        v16h b_v10 = *(const v16h*)(vb + (size_t)(n + 16) * S_ + jv0);
        v16h b_v11 = *(const v16h*)(vb + (size_t)(n + 16) * S_ + jv1);

        // --- rescale accumulators (per-row alpha is in-lane) and accumulate P*V
        #pragma unroll
        for (int r = 0; r < 8; ++r) { o0[r] *= alpha[r]; o1[r] *= alpha[r]; }
        o0 = __builtin_amdgcn_wmma_f32_16x16x32_f16(false, a_p0, false, b_v00,
                                                    (short)0, o0, false, false);
        o0 = __builtin_amdgcn_wmma_f32_16x16x32_f16(false, a_p1, false, b_v01,
                                                    (short)0, o0, false, false);
        o1 = __builtin_amdgcn_wmma_f32_16x16x32_f16(false, a_p0, false, b_v10,
                                                    (short)0, o1, false, false);
        o1 = __builtin_amdgcn_wmma_f32_16x16x32_f16(false, a_p1, false, b_v11,
                                                    (short)0, o1, false, false);

        // prefetch next K chunk into caches
        if (j0 + 64 <= i0)
            __builtin_prefetch(kb + (size_t)(j0 + 64 + n) * D_, 0, 1);
    }

    // --- epilogue: out[b, s, h*32 + d] = O / l
    const int b = bh >> 2, h = bh & 3;
    #pragma unroll
    for (int r = 0; r < 8; ++r) {
        int srow = i0 + 8 * half + r;
        float inv = 1.0f / lstat[r];
        float* orow = out + ((size_t)b * S_ + srow) * HID_ + h * D_;
        orow[n]      = o0[r] * inv;
        orow[16 + n] = o1[r] * inv;
    }
}

// ---------------------------------------------------------------------------
extern "C" void kernel_launch(void* const* d_in, const int* in_sizes, int n_in,
                              void* d_out, int out_size, void* d_ws, size_t ws_size,
                              hipStream_t stream) {
    const float* x  = (const float*)d_in[0];
    const float* Wq = (const float*)d_in[1];
    const float* bq = (const float*)d_in[2];
    const float* Wk = (const float*)d_in[3];
    const float* bk = (const float*)d_in[4];
    const float* Wv = (const float*)d_in[5];
    const float* bv = (const float*)d_in[6];
    const float* Er = (const float*)d_in[7];
    float* out = (float*)d_out;

    const size_t QKV_ELEMS = (size_t)B_ * H_ * S_ * D_;   // 2,097,152
    f16* q16  = (f16*)d_ws;
    f16* k16  = q16  + QKV_ELEMS;
    f16* vt16 = k16  + QKV_ELEMS;
    f16* er16 = vt16 + QKV_ELEMS;

    rga_er_cvt<<<(S_ * D_ + 255) / 256, 256, 0, stream>>>(Er, er16, S_ * D_);

    rga_qkv_proj<<<(B_ * S_) / 8, 128, 0, stream>>>(x, Wq, bq, Wk, bk, Wv, bv,
                                                    q16, k16, vt16);

    // one wave per 16-query tile; 8 waves (tiles) per 256-thread block
    rga_flash<<<(B_ * H_ * (S_ / 16)) / 8, 256, 0, stream>>>(q16, k16, vt16, er16, out);
}
